// ParallelEnrichmentBlock_42210938585193
// MI455X (gfx1250) — compile-verified, matrix operand order checked by reading, not run on hardware
//
#include <hip/hip_runtime.h>

// ---------------------------------------------------------------------------
// Problem constants (from reference): B=4, T=2048, D=1024, E=4, WR=128, H=4096
// ---------------------------------------------------------------------------
#define NTOK   8192          // B*T
#define DMODEL 1024
#define KFUSE  1152          // D + WR
#define HDIM   4096
#define NPATH  5             // 1 + E
#define ZDIM   5120          // NPATH * D

typedef __attribute__((ext_vector_type(16))) __bf16 v16bf;
typedef __attribute__((ext_vector_type(8)))  float  v8f;
typedef __attribute__((ext_vector_type(4)))  int    i32x4;

union BFrag {
    unsigned int u[8];
    v16bf        v;
};

__device__ __forceinline__ unsigned short f32_to_bf16(float f) {
    unsigned int u = __float_as_uint(f);
    unsigned int lsb = (u >> 16) & 1u;
    u += 0x7fffu + lsb;                      // round-to-nearest-even
    return (unsigned short)(u >> 16);
}

__device__ __forceinline__ float gelu_t(float x) {
    const float k0 = 0.7978845608028654f;    // sqrt(2/pi)
    const float k1 = 0.044715f;
    float t = tanhf(k0 * (x + k1 * x * x * x));
    return 0.5f * x * (1.0f + t);
}

// ---------------------------------------------------------------------------
// Async global->LDS (CDNA5 GLOBAL_LOAD_ASYNC_TO_LDS_B128, ASYNCcnt-tracked).
// Guarded: falls back to a synchronous copy if the builtin is absent.
// Builtin signature (from hipcc diagnostic): (AS1 int4*, AS3 int4*, imm, imm)
// ---------------------------------------------------------------------------
#if __has_builtin(__builtin_amdgcn_global_load_async_to_lds_b128)
#define HAVE_ASYNC_LDS 1
typedef __attribute__((address_space(1))) i32x4 g_i32x4;
typedef __attribute__((address_space(3))) i32x4 l_i32x4;
#else
#define HAVE_ASYNC_LDS 0
#endif

__device__ __forceinline__ void async_copy_b128(const unsigned short* gsrc,
                                                unsigned short* ldst) {
#if HAVE_ASYNC_LDS
    __builtin_amdgcn_global_load_async_to_lds_b128(
        (g_i32x4*)gsrc, (l_i32x4*)ldst, 0, 0);
#else
    *(uint4*)ldst = *(const uint4*)gsrc;
#endif
}

template <int N>
__device__ __forceinline__ void async_wait_n() {
#if HAVE_ASYNC_LDS
#if __has_builtin(__builtin_amdgcn_s_wait_asynccnt)
    __builtin_amdgcn_s_wait_asynccnt(N);
#else
    asm volatile("s_wait_asynccnt %0" :: "i"(N) : "memory");
#endif
#endif
}

// ---------------------------------------------------------------------------
// Kernel: tiled transpose f32 [K,N] -> bf16 [N,K], coalesced both directions
// grid: (N/32, K/32), 256 threads (32x8)
// ---------------------------------------------------------------------------
__global__ __launch_bounds__(256)
void transpose_cast_bf16(const float* __restrict__ w,
                         unsigned short* __restrict__ wt,
                         int K, int N) {
    __shared__ float tile[32][33];
    int nb = blockIdx.x * 32;
    int kb = blockIdx.y * 32;
    int tx = threadIdx.x & 31;
    int ty = threadIdx.x >> 5;           // 0..7
#pragma unroll
    for (int i = 0; i < 32; i += 8)
        tile[ty + i][tx] = w[(size_t)(kb + ty + i) * N + nb + tx];
    __syncthreads();
#pragma unroll
    for (int i = 0; i < 32; i += 8)
        wt[(size_t)(nb + ty + i) * K + kb + tx] = f32_to_bf16(tile[tx][ty + i]);
}

// ---------------------------------------------------------------------------
// Kernel: LN(cat(x, r_feat)) -> bf16 A matrix [NTOK, KFUSE]
// One wave (32 lanes) per token, 8 tokens per 256-thread block.
// ---------------------------------------------------------------------------
__global__ __launch_bounds__(256)
void ln_fuse_kernel(const float* __restrict__ x, const float* __restrict__ rf,
                    const float* __restrict__ g, const float* __restrict__ b,
                    unsigned short* __restrict__ aln) {
    int wave = threadIdx.x >> 5;
    int lane = threadIdx.x & 31;
    int tok  = blockIdx.x * 8 + wave;

    const float* xrow = x  + (size_t)tok * DMODEL;
    const float* rrow = rf + (size_t)tok * 128;

    float v[36];
    float s = 0.f, sq = 0.f;
#pragma unroll
    for (int j = 0; j < 36; ++j) {
        int f = j * 32 + lane;
        float val = (f < DMODEL) ? xrow[f] : rrow[f - DMODEL];
        v[j] = val;
        s  += val;
        sq += val * val;
    }
#pragma unroll
    for (int off = 16; off > 0; off >>= 1) {
        s  += __shfl_xor(s,  off, 32);
        sq += __shfl_xor(sq, off, 32);
    }
    float mean = s * (1.0f / KFUSE);
    float var  = sq * (1.0f / KFUSE) - mean * mean;
    float rs   = rsqrtf(var + 1e-5f);

    unsigned short* orow = aln + (size_t)tok * KFUSE;
#pragma unroll
    for (int j = 0; j < 36; ++j) {
        int f = j * 32 + lane;
        orow[f] = f32_to_bf16((v[j] - mean) * rs * g[f] + b[f]);
    }
}

// ---------------------------------------------------------------------------
// WMMA bf16 GEMM:  C[M,N] = epilogue(A[M,K] * Bt[N,K]^T)
//   Block tile 128(M) x 256(N), BK=32, 256 threads = 8 waves (2 x 4),
//   each wave computes a 64x64 slab = 4x4 tiles of v_wmma_f32_16x16x32_bf16
//   (16 WMMAs per 16 LDS b128 fragment loads per K-step).
//   Double-buffered LDS fed by async global->LDS b128 loads.
// EPI==1: C = gelu(acc + bias)        -> bf16
// EPI==2: C = acc + bias + addend     -> f32
// ---------------------------------------------------------------------------
#define LSTR 40   // LDS row stride in bf16 elements (32 + 8 pad), 80B, 16B-aligned

template <int EPI>
__global__ __launch_bounds__(256)
void wmma_gemm(const unsigned short* __restrict__ A,
               const unsigned short* __restrict__ Bt,
               int K, int N,
               const float* __restrict__ bias,
               const float* __restrict__ addend,
               void* __restrict__ Cout) {
    __shared__ unsigned short shA[2][128 * LSTR];   // 2 x 10,240 B
    __shared__ unsigned short shB[2][256 * LSTR];   // 2 x 20,480 B

    int tid  = threadIdx.x;
    int wave = tid >> 5;
    int lane = tid & 31;
    int wm   = wave & 1;       // 0..1 -> 64-row slab
    int wn   = wave >> 1;      // 0..3 -> 64-col slab
    int m0   = blockIdx.x * 128;
    int n0   = blockIdx.y * 256;

    v8f acc[4][4] = {};

    int khalf = (lane >= 16) ? 8 : 0;   // ISA 16-bit A/B-matrix K striping
    int lmod  = lane & 15;

    // per-thread chunk coords for tile fills (b128 = 8 bf16)
    int crow = tid >> 2;            // 0..63
    int ccol = (tid & 3) * 8;       // 0,8,16,24

    auto issue_tile = [&](int nb, int kt2) {
        int k0 = kt2 * 32;
#pragma unroll
        for (int p = 0; p < 2; ++p) {    // A: 128 rows
            int row = crow + p * 64;
            async_copy_b128(&A[(size_t)(m0 + row) * K + k0 + ccol],
                            &shA[nb][row * LSTR + ccol]);
        }
#pragma unroll
        for (int p = 0; p < 4; ++p) {    // B: 256 rows
            int row = crow + p * 64;
            async_copy_b128(&Bt[(size_t)(n0 + row) * K + k0 + ccol],
                            &shB[nb][row * LSTR + ccol]);
        }
    };

    int kt_iters = K / 32;
    issue_tile(0, 0);

    for (int kt = 0; kt < kt_iters; ++kt) {
        int cur = kt & 1;
        if (kt + 1 < kt_iters) {
            issue_tile(1 - cur, kt + 1);   // prefetch next tile (6 async loads)
            async_wait_n<6>();             // current tile's loads retired
        } else {
            async_wait_n<0>();
        }
        __syncthreads();

        BFrag a[4], bfr[4];
#pragma unroll
        for (int mi = 0; mi < 4; ++mi) {
            const unsigned short* p = &shA[cur][(wm * 64 + mi * 16 + lmod) * LSTR + khalf];
            uint4 t0 = *(const uint4*)p;         // K = khalf .. khalf+7
            uint4 t1 = *(const uint4*)(p + 16);  // K = 16+khalf ..
            a[mi].u[0] = t0.x; a[mi].u[1] = t0.y; a[mi].u[2] = t0.z; a[mi].u[3] = t0.w;
            a[mi].u[4] = t1.x; a[mi].u[5] = t1.y; a[mi].u[6] = t1.z; a[mi].u[7] = t1.w;
        }
#pragma unroll
        for (int ni = 0; ni < 4; ++ni) {
            const unsigned short* p = &shB[cur][(wn * 64 + ni * 16 + lmod) * LSTR + khalf];
            uint4 t0 = *(const uint4*)p;
            uint4 t1 = *(const uint4*)(p + 16);
            bfr[ni].u[0] = t0.x; bfr[ni].u[1] = t0.y; bfr[ni].u[2] = t0.z; bfr[ni].u[3] = t0.w;
            bfr[ni].u[4] = t1.x; bfr[ni].u[5] = t1.y; bfr[ni].u[6] = t1.z; bfr[ni].u[7] = t1.w;
        }

#pragma unroll
        for (int mi = 0; mi < 4; ++mi)
#pragma unroll
            for (int ni = 0; ni < 4; ++ni)
                acc[mi][ni] = __builtin_amdgcn_wmma_f32_16x16x32_bf16(
                    false, a[mi].v, false, bfr[ni].v,
                    (short)0, acc[mi][ni], false, false);

        __syncthreads();
    }

    // epilogue: C element (r, lane) per ISA 16x16 f32 C/D layout
    int mhalf = (lane >= 16) ? 8 : 0;
#pragma unroll
    for (int mi = 0; mi < 4; ++mi) {
#pragma unroll
        for (int ni = 0; ni < 4; ++ni) {
            int n = n0 + wn * 64 + ni * 16 + lmod;
            float bn = bias[n];
#pragma unroll
            for (int r = 0; r < 8; ++r) {
                int m = m0 + wm * 64 + mi * 16 + r + mhalf;
                float v = acc[mi][ni][r];
                if (EPI == 1) {
                    v = gelu_t(v + bn);
                    ((unsigned short*)Cout)[(size_t)m * N + n] = f32_to_bf16(v);
                } else {
                    v = v + bn + addend[(size_t)m * N + n];
                    ((float*)Cout)[(size_t)m * N + n] = v;
                }
            }
        }
    }
}

// ---------------------------------------------------------------------------
// Kernel: path gating. One 256-thread block per token.
// ---------------------------------------------------------------------------
__global__ __launch_bounds__(256)
void gate_kernel(const float* __restrict__ vt2, const float* __restrict__ dts,
                 const float* __restrict__ g, const float* __restrict__ b,
                 const float* __restrict__ gw, const float* __restrict__ gb,
                 float* __restrict__ out, float* __restrict__ gmem) {
    int tok = blockIdx.x;
    int tid = threadIdx.x;

    __shared__ float zsh[ZDIM];
    __shared__ float red[256];
    __shared__ float pish[8];

    float zr[20];
    float s = 0.f, sq = 0.f;
#pragma unroll
    for (int j = 0; j < 20; ++j) {
        int f = j * 256 + tid;
        int p = f >> 10;
        int d = f & 1023;
        float val = (p == 0)
            ? vt2[(size_t)tok * DMODEL + d]
            : dts[((size_t)(p - 1) * NTOK + tok) * DMODEL + d];
        zsh[f] = val;
        zr[j]  = val;
        s  += val;
        sq += val * val;
    }
    __syncthreads();

    red[tid] = s; __syncthreads();
    for (int off = 128; off > 0; off >>= 1) { if (tid < off) red[tid] += red[tid + off]; __syncthreads(); }
    float mean = red[0] * (1.0f / ZDIM);
    __syncthreads();
    red[tid] = sq; __syncthreads();
    for (int off = 128; off > 0; off >>= 1) { if (tid < off) red[tid] += red[tid + off]; __syncthreads(); }
    float var = red[0] * (1.0f / ZDIM) - mean * mean;
    float rs  = rsqrtf(var + 1e-5f);

    float lp[NPATH] = {0.f, 0.f, 0.f, 0.f, 0.f};
#pragma unroll
    for (int j = 0; j < 20; ++j) {
        int f = j * 256 + tid;
        float zn = (zr[j] - mean) * rs * g[f] + b[f];
#pragma unroll
        for (int p = 0; p < NPATH; ++p) lp[p] += zn * gw[f * NPATH + p];
    }
    for (int p = 0; p < NPATH; ++p) {
        __syncthreads();
        red[tid] = lp[p]; __syncthreads();
        for (int off = 128; off > 0; off >>= 1) { if (tid < off) red[tid] += red[tid + off]; __syncthreads(); }
        if (tid == 0) pish[p] = red[0] + gb[p];
    }
    __syncthreads();
    if (tid == 0) {
        float mx = pish[0];
        for (int p = 1; p < NPATH; ++p) mx = fmaxf(mx, pish[p]);
        float e[NPATH], ssum = 0.f;
        for (int p = 0; p < NPATH; ++p) { e[p] = __expf(pish[p] - mx); ssum += e[p]; }
        float inv = 1.0f / ssum;
        for (int p = 0; p < NPATH; ++p) pish[p] = e[p] * inv;
        gmem[tok] = pish[1] + pish[2] + pish[3] + pish[4];
    }
    __syncthreads();
    float p0 = pish[0], p1 = pish[1], p2 = pish[2], p3 = pish[3], p4 = pish[4];
#pragma unroll
    for (int j = 0; j < 4; ++j) {
        int d = j * 256 + tid;
        out[(size_t)tok * DMODEL + d] =
            p0 * zsh[d] + p1 * zsh[1024 + d] + p2 * zsh[2048 + d] +
            p3 * zsh[3072 + d] + p4 * zsh[4096 + d];
    }
}

// ---------------------------------------------------------------------------
// Host launcher
// ---------------------------------------------------------------------------
extern "C" void kernel_launch(void* const* d_in, const int* in_sizes, int n_in,
                              void* d_out, int out_size, void* d_ws, size_t ws_size,
                              hipStream_t stream) {
    const float* x        = (const float*)d_in[0];
    const float* vt       = (const float*)d_in[1];
    const float* dts      = (const float*)d_in[2];
    const float* r_feat   = (const float*)d_in[3];
    const float* fuse_g   = (const float*)d_in[4];
    const float* fuse_b   = (const float*)d_in[5];
    const float* fuse_w1  = (const float*)d_in[6];
    const float* fuse_b1  = (const float*)d_in[7];
    const float* fuse_w2  = (const float*)d_in[8];
    const float* fuse_b2  = (const float*)d_in[9];
    const float* pg_g     = (const float*)d_in[10];
    const float* pg_b     = (const float*)d_in[11];
    const float* gate_w   = (const float*)d_in[12];
    const float* gate_b   = (const float*)d_in[13];

    float* out  = (float*)d_out;                 // [NTOK, D]
    float* gmem = out + (size_t)NTOK * DMODEL;   // [NTOK, 1]

    // workspace layout (bytes, 16B-aligned)
    char* ws = (char*)d_ws;
    unsigned short* aln  = (unsigned short*)(ws + 0);            // 18,874,368  A bf16 [NTOK,1152]
    unsigned short* w1t  = (unsigned short*)(ws + 18874368);     //  9,437,184  W1^T bf16 [4096,1152]
    unsigned short* w2t  = (unsigned short*)(ws + 28311552);     //  8,388,608  W2^T bf16 [1024,4096]
    unsigned short* hbuf = (unsigned short*)(ws + 36700160);     // 67,108,864  h bf16 [NTOK,4096]
    float*          vt2  = (float*)        (ws + 103809024);     // 33,554,432  vt2 f32 [NTOK,1024]
    // total: 137,363,456 bytes

    // 1) weights -> bf16 transposed (tiled, coalesced)
    {
        dim3 g1(HDIM / 32, KFUSE / 32);
        transpose_cast_bf16<<<g1, 256, 0, stream>>>(fuse_w1, w1t, KFUSE, HDIM);
        dim3 g2(DMODEL / 32, HDIM / 32);
        transpose_cast_bf16<<<g2, 256, 0, stream>>>(fuse_w2, w2t, HDIM, DMODEL);
    }

    // 2) LN(cat(x, r_feat)) -> bf16
    ln_fuse_kernel<<<NTOK / 8, 256, 0, stream>>>(x, r_feat, fuse_g, fuse_b, aln);

    // 3) h = gelu(aln @ W1 + b1)   [NTOK, HDIM] bf16
    {
        dim3 grid(NTOK / 128, HDIM / 256);
        wmma_gemm<1><<<grid, 256, 0, stream>>>(aln, w1t, KFUSE, HDIM, fuse_b1, nullptr, (void*)hbuf);
    }

    // 4) vt2 = vt + h @ W2 + b2    [NTOK, DMODEL] f32
    {
        dim3 grid(NTOK / 128, DMODEL / 256);
        wmma_gemm<2><<<grid, 256, 0, stream>>>(hbuf, w2t, HDIM, DMODEL, fuse_b2, vt, (void*)vt2);
    }

    // 5) gating + combine
    gate_kernel<<<NTOK, 256, 0, stream>>>(vt2, dts, pg_g, pg_b, gate_w, gate_b, out, gmem);
}
